// Linear_mvm_47811575939413
// MI455X (gfx1250) — 8-bit matrix operand order repaired and checked by reading, compiled, not run on hardware
//
#include <hip/hip_runtime.h>

typedef __attribute__((ext_vector_type(8))) int v8i;

#define A_PRE_OFF   0
#define B_PRE_OFF   ((size_t)4 * 1024 * 1024)
#define PART_OFF    ((size_t)20 * 1024 * 1024)

__global__ void prep_bits_kernel(const float* __restrict__ x,
                                 unsigned char* __restrict__ A_pre) {
  int idx = blockIdx.x * blockDim.x + threadIdx.x;
  int b = idx >> 10, i = idx & 1023;
  float r = rintf(x[idx] * 4096.0f);
  r = fminf(fmaxf(r, -32768.0f), 32767.0f);
  unsigned int xu = ((unsigned int)(int)r) & 0xFFFFu;
  int bt = b >> 4, m = b & 15;
  int xr = i >> 6, K = i & 63;
  int lane = (((K >> 4) & 1) << 4) | m;
  int vr   = ((K >> 5) << 2) | ((K >> 2) & 3);
  int byt  = K & 3;
  unsigned int base = (unsigned int)((bt * 16 + xr) * 1024 + lane * 32 + vr * 4 + byt);
#pragma unroll
  for (int t = 0; t < 16; ++t)
    A_pre[(unsigned int)t * (16u * 16u * 1024u) + base] =
        (unsigned char)((xu >> t) & 1u);
}

__global__ void prep_weight_kernel(const float* __restrict__ weight,
                                   unsigned char* __restrict__ B_pre) {
  int idx = blockIdx.x * blockDim.x + threadIdx.x;
  int o = idx >> 10, i = idx & 1023;
  float w = weight[idx];
  float rp = fminf(fmaxf(rintf(fmaxf(w, 0.0f) * 4096.0f), 0.0f), 65535.0f);
  float rn = fminf(fmaxf(rintf(fmaxf(-w, 0.0f) * 4096.0f), 0.0f), 65535.0f);
  unsigned int pi = (unsigned int)rp, ni = (unsigned int)rn;
  int xr = i >> 6, K = i & 63;
  int ot = o >> 4, n = o & 15;
  int lane = (((K >> 4) & 1) << 4) | n;
  int vr   = ((K >> 5) << 2) | ((K >> 2) & 3);
  int byt  = K & 3;
  unsigned int base = (unsigned int)(((xr * 64 + ot) * 1024) + lane * 32 + vr * 4 + byt);
#pragma unroll
  for (int s = 0; s < 8; ++s) {
    int sh = 2 * (7 - s);
    B_pre[(unsigned int)s       * (16u * 64u * 1024u) + base] = (unsigned char)((pi >> sh) & 3u);
    B_pre[(unsigned int)(s + 8) * (16u * 64u * 1024u) + base] = (unsigned char)((ni >> sh) & 3u);
  }
}

__global__ void __launch_bounds__(256, 1)
mvm_wmma_kernel(const unsigned char* __restrict__ A_pre,
                const unsigned char* __restrict__ B_pre,
                float* __restrict__ partial) {
  const int lane = threadIdx.x & 31;
  const int wid  = (blockIdx.x * blockDim.x + threadIdx.x) >> 5;
  const int ot   = wid & 63;
  const int xrg  = (wid >> 6) & 3;
  const int bt   = wid >> 8;

  const float step = 192.0f / 255.0f;
  float facc[8] = {0.f, 0.f, 0.f, 0.f, 0.f, 0.f, 0.f, 0.f};

  const int xr_end = xrg * 4 + 4;
#pragma unroll 1
  for (int xr = xrg * 4; xr < xr_end; ++xr) {
    v8i Breg[16];
#pragma unroll
    for (int u = 0; u < 16; ++u)
      Breg[u] = *(const v8i*)(B_pre +
                 ((size_t)(u * 16 + xr) * 64 + ot) * 1024 + (size_t)lane * 32);

    const unsigned char* Ap =
        A_pre + (size_t)(bt * 16 + xr) * 1024 + (size_t)lane * 32;

    if (xr + 1 < xr_end)
      __builtin_prefetch(B_pre + ((size_t)(xr + 1) * 64 + ot) * 1024 + (size_t)lane * 32, 0, 2);

#pragma unroll 1
    for (int t = 0; t < 16; ++t) {
      v8i Av = *(const v8i*)(Ap + (size_t)t * (256 * 1024));
      float wtb = (float)(1 << t);
      if (t == 15) wtb = -wtb;
#pragma unroll
      for (int u = 0; u < 16; ++u) {
        v8i c = {0, 0, 0, 0, 0, 0, 0, 0};
        c = __builtin_amdgcn_wmma_i32_16x16x64_iu8(
                false, Av, false, Breg[u], c, false, false);
        const int s = u & 7;
        const float sgn = (u < 8) ? 1.0f : -1.0f;
        const float wt  = wtb * sgn * (float)(1 << (2 * (7 - s))) * step;
#pragma unroll
        for (int k = 0; k < 8; ++k) {
          float q = rintf(fmaf((float)c[k], 1.328125f, -0.0009765625f));
          facc[k] = fmaf(q, wt, facc[k]);
        }
      }
    }
  }

  const int n  = lane & 15;
  const int mh = (lane >> 4) * 8;
#pragma unroll
  for (int k = 0; k < 8; ++k) {
    int b = bt * 16 + mh + k;
    partial[((size_t)xrg * 256 + b) * 1024 + (size_t)ot * 16 + n] = facc[k];
  }
}

__global__ void finalize_kernel(const float* __restrict__ partial,
                                const float* __restrict__ bias,
                                float* __restrict__ out) {
  int idx = blockIdx.x * blockDim.x + threadIdx.x;
  float s = partial[idx] + partial[262144 + idx] +
            partial[2 * 262144 + idx] + partial[3 * 262144 + idx];
  s *= 0x1p-24f;
  float q = fminf(fmaxf(rintf(s * 4096.0f), -32768.0f), 32767.0f);
  out[idx] = q * 0x1p-12f + bias[idx & 1023];
}

extern "C" void kernel_launch(void* const* d_in, const int* in_sizes, int n_in,
                              void* d_out, int out_size, void* d_ws, size_t ws_size,
                              hipStream_t stream) {
  const float* x      = (const float*)d_in[0];
  const float* weight = (const float*)d_in[1];
  const float* bias   = (const float*)d_in[2];
  float* out = (float*)d_out;

  unsigned char* ws    = (unsigned char*)d_ws;
  unsigned char* A_pre = ws + A_PRE_OFF;
  unsigned char* B_pre = ws + B_PRE_OFF;
  float*         part  = (float*)(ws + PART_OFF);

  prep_bits_kernel  <<<1024, 256, 0, stream>>>(x, A_pre);
  prep_weight_kernel<<<4096, 256, 0, stream>>>(weight, B_pre);
  mvm_wmma_kernel   <<< 512, 256, 0, stream>>>(A_pre, B_pre, part);
  finalize_kernel   <<<1024, 256, 0, stream>>>(part, bias, out);
}
